// RWKV7PoSTMixer_42125039239449
// MI455X (gfx1250) — matrix operand order repaired and checked by reading, not compiled, not run on hardware
//
#include <hip/hip_runtime.h>
#include <math.h>

// ---------------- problem constants ----------------
constexpr int  Bsz  = 4;
constexpr int  Tseq = 2048;
constexpr int  Cdim = 1024;
constexpr int  Hn   = 16;
constexpr int  HDn  = 64;
constexpr int  DW_  = 64;
constexpr int  DA_  = 64;
constexpr int  DG_  = 128;
constexpr long long BTll = (long long)Bsz * Tseq;   // 8192 rows
constexpr float EPS_ = 64e-5f;

// ---------------- CDNA5 vector types ----------------
typedef __attribute__((ext_vector_type(16))) __bf16 v16bf;
typedef __attribute__((ext_vector_type(8)))  __bf16 v8bf;
typedef __attribute__((ext_vector_type(8)))  float  v8f;
typedef __attribute__((ext_vector_type(2)))  float  v2f;

// =====================================================================
// f32 -> bf16 convert (weights)
// =====================================================================
__global__ void cvt_f32_bf16(const float* __restrict__ src,
                             __bf16* __restrict__ dst, long long n) {
    long long i = (long long)blockIdx.x * blockDim.x + threadIdx.x;
    if (i < n) dst[i] = (__bf16)src[i];
}

// =====================================================================
// token shift: xx = x[t-1]-x[t];  x? = x + xx*coef
// outputs: bf16 xr/xk/xv (GEMM inputs), f32 xw/xa/xg (low-rank inputs)
// =====================================================================
__global__ void token_shift_kernel(const float* __restrict__ x,
    const float* __restrict__ cr, const float* __restrict__ cw,
    const float* __restrict__ ck, const float* __restrict__ cv,
    const float* __restrict__ ca, const float* __restrict__ cg,
    __bf16* __restrict__ xr_bf, __bf16* __restrict__ xk_bf,
    __bf16* __restrict__ xv_bf,
    float* __restrict__ xw_f, float* __restrict__ xa_f,
    float* __restrict__ xg_f)
{
    long long idx = (long long)blockIdx.x * blockDim.x + threadIdx.x;
    long long n = BTll * Cdim;
    if (idx >= n) return;
    int c = (int)(idx % Cdim);
    long long bt = idx / Cdim;
    int t = (int)(bt % Tseq);
    float xc = x[idx];
    float xp = (t == 0) ? 0.f : x[idx - Cdim];
    float xx = xp - xc;
    xr_bf[idx] = (__bf16)(xc + xx * cr[c]);
    xk_bf[idx] = (__bf16)(xc + xx * ck[c]);
    xv_bf[idx] = (__bf16)(xc + xx * cv[c]);
    xw_f[idx]  = xc + xx * cw[c];
    xa_f[idx]  = xc + xx * ca[c];
    xg_f[idx]  = xc + xx * cg[c];
}

// =====================================================================
// WMMA bf16 GEMM:  Cout[M,N] f32 = A[M,K] bf16 (row-major) @ W[K,N] bf16
// Block tile 256x64, 8 waves (4x2), wave tile 64x32 (4x2 wmma tiles),
// K step 32.  A staged row-major in LDS; W staged transposed (N-major)
// so per-lane fragment loads are contiguous 16B reads (ISA layouts).
// =====================================================================
#define TILE_M 256
#define TILE_N 64
#define TILE_K 32
#define LDS_STRIDE 40   // 32 + 8 bf16 pad

__global__ __launch_bounds__(256)
void wmma_gemm_bf16(const __bf16* __restrict__ A,
                    const __bf16* __restrict__ Wm,
                    float* __restrict__ Cout,
                    int M, int N, int K)
{
    __shared__ __bf16 sA [TILE_M * LDS_STRIDE];   // 20 KB
    __shared__ __bf16 sBt[TILE_N * LDS_STRIDE];   //  5 KB

    const int tid  = threadIdx.x;
    const int lane = tid & 31;
    const int wid  = tid >> 5;
    const int wm   = wid >> 1;       // 0..3  -> 64-row strip
    const int wn   = wid & 1;        // 0..1  -> 32-col strip

    const int bm = blockIdx.x * TILE_M;
    const int bn = blockIdx.y * TILE_N;

    v8f acc[4][2] = {};

    // per-lane fragment offsets (ISA §7.12.2 layouts)
    const int a_klo = (lane < 16) ? 0 : 8;    // A: elems 0..7  -> K a_klo..+7
    const int a_khi = a_klo + 16;             //    elems 8..15 -> K a_khi..+7
    const int b_k0  = (lane < 16) ? 0 : 8;    // B: elems 0..7 -> K b_k0..+7, 8..15 -> K b_k0+16..+7
    const int frow  = lane & 15;

    for (int kb = 0; kb < K; kb += TILE_K) {
        // ---- stage A slab 256x32 (each thread: one row, 32 bf16) ----
        {
            const __bf16* gp = A + (size_t)(bm + tid) * K + kb;
            v8bf a0 = *(const v8bf*)gp;
            v8bf a1 = *(const v8bf*)(gp + 8);
            v8bf a2 = *(const v8bf*)(gp + 16);
            v8bf a3 = *(const v8bf*)(gp + 24);
            if (kb + TILE_K < K) __builtin_prefetch(gp + TILE_K, 0, 3);
            __bf16* sp = &sA[tid * LDS_STRIDE];
            *(v8bf*)(sp)      = a0;
            *(v8bf*)(sp + 8)  = a1;
            *(v8bf*)(sp + 16) = a2;
            *(v8bf*)(sp + 24) = a3;
        }
        // ---- stage W slab 32x64 transposed -> sBt[n][k] ----
        {
            int idx  = tid * 8;
            int krow = idx >> 6;       // 0..31
            int ncol = idx & 63;       // multiple of 8
            const __bf16* gp = Wm + (size_t)(kb + krow) * N + bn + ncol;
            v8bf bv = *(const v8bf*)gp;
            if (kb + TILE_K < K) __builtin_prefetch(gp + (size_t)TILE_K * N, 0, 3);
            #pragma unroll
            for (int e = 0; e < 8; ++e)
                sBt[(ncol + e) * LDS_STRIDE + krow] = bv[e];
        }
        __syncthreads();

        // ---- B fragments (shared across i) ----
        v16bf bfrag[2];
        #pragma unroll
        for (int j = 0; j < 2; ++j) {
            const __bf16* bp = &sBt[(wn * 32 + j * 16 + frow) * LDS_STRIDE];
            v8bf blo = *(const v8bf*)(bp + b_k0);
            v8bf bhi = *(const v8bf*)(bp + b_k0 + 16);
            #pragma unroll
            for (int e = 0; e < 8; ++e) { bfrag[j][e] = blo[e]; bfrag[j][8 + e] = bhi[e]; }
        }
        // ---- A fragments + 8 WMMAs ----
        #pragma unroll
        for (int i = 0; i < 4; ++i) {
            const __bf16* ap = &sA[(wm * 64 + i * 16 + frow) * LDS_STRIDE];
            v8bf alo = *(const v8bf*)(ap + a_klo);
            v8bf ahi = *(const v8bf*)(ap + a_khi);
            v16bf afrag;
            #pragma unroll
            for (int e = 0; e < 8; ++e) { afrag[e] = alo[e]; afrag[8 + e] = ahi[e]; }
            #pragma unroll
            for (int j = 0; j < 2; ++j) {
                acc[i][j] = __builtin_amdgcn_wmma_f32_16x16x32_bf16(
                    false, afrag, false, bfrag[j], (short)0, acc[i][j],
                    false, false);
            }
        }
        __syncthreads();
    }

    // ---- store 4x2 tiles (C/D layout: vgpr r -> M = r + (lane<16?0:8)) ----
    const int mh = (lane < 16) ? 0 : 8;
    #pragma unroll
    for (int i = 0; i < 4; ++i)
        #pragma unroll
        for (int j = 0; j < 2; ++j)
            #pragma unroll
            for (int rg = 0; rg < 8; ++rg) {
                int row = bm + wm * 64 + i * 16 + mh + rg;
                int col = bn + wn * 32 + j * 16 + frow;
                Cout[(size_t)row * N + col] = acc[i][j][rg];
            }
}

// =====================================================================
// Full-precision WMMA GEMM on V_WMMA_F32_16X16X4_F32 for low-rank chains.
// One wave per 16x16 output tile; K stepped by 4.
//   A fragment (16x4 f32): lanes 0-15 hold K=k,k+1; lanes 16-31 K=k+2,k+3
//   B fragment (4x16 f32): column = lane&15, same K split.
// Epilogue modes: 0 none | 1 tanh | 2 sigmoid | 3 sigmoid(b0+s)
//                 4 decay: exp(-exp(-softplus(-(b0+s)) - 0.5))
// =====================================================================
__global__ __launch_bounds__(32)
void wmma_gemm_f32(const float* __restrict__ A, const float* __restrict__ Bm,
                   const float* __restrict__ bias, float* __restrict__ Out,
                   int M, int K, int N, int mode)
{
    const int tm   = blockIdx.x * 16;
    const int tn   = blockIdx.y * 16;
    const int lane = threadIdx.x;
    const int half = lane >> 4;     // 0: K=k..k+1, 1: K=k+2..k+3
    const int l15  = lane & 15;

    const float* arow = A + (size_t)(tm + l15) * K;
    v8f acc = {};
    for (int k = 0; k < K; k += 4) {
        int ka = k + half * 2;
        v2f af, bf;
        af[0] = arow[ka];
        af[1] = arow[ka + 1];
        bf[0] = Bm[(size_t)ka * N + tn + l15];
        bf[1] = Bm[(size_t)(ka + 1) * N + tn + l15];
        acc = __builtin_amdgcn_wmma_f32_16x16x4_f32(
            false, af, false, bf, (short)0, acc, false, false);
    }

    const int mh  = half * 8;
    const int col = tn + l15;
    #pragma unroll
    for (int rg = 0; rg < 8; ++rg) {
        int row = tm + mh + rg;
        float s = acc[rg];
        if (mode == 1)      s = tanhf(s);
        else if (mode == 2) s = 1.f / (1.f + expf(-s));
        else if (mode == 3) s = 1.f / (1.f + expf(-(bias[col] + s)));
        else if (mode == 4) {
            float wl = bias[col] + s;
            float w  = -log1pf(expf(-wl)) - 0.5f;    // -softplus(-wl) - 0.5
            s = expf(-expf(w));                      // decay in (0,1)
        }
        Out[(size_t)row * N + col] = s;
    }
}

// =====================================================================
// kk = normalize(k*k_k) per head; kmod = k*(1+(a-1)*k_a)  (in-place over k)
// =====================================================================
__global__ void prep_kernel(const float* __restrict__ k_in,
                            const float* __restrict__ a_in,
                            const float* __restrict__ k_k,
                            const float* __restrict__ k_a,
                            float* __restrict__ kk_out,
                            float* __restrict__ k_out)
{
    long long idx = (long long)blockIdx.x * blockDim.x + threadIdx.x;
    if (idx >= BTll * Hn) return;
    size_t base = (size_t)idx * HDn;
    int h = (int)(idx % Hn);
    const float* kkc = k_k + h * HDn;
    const float* kac = k_a + h * HDn;
    float ss = 0.f;
    for (int j = 0; j < HDn; ++j) { float t = k_in[base + j] * kkc[j]; ss = fmaf(t, t, ss); }
    float inv = 1.f / fmaxf(sqrtf(ss), 1e-12f);
    for (int j = 0; j < HDn; ++j) {
        float kv = k_in[base + j];
        float av = a_in[base + j];
        kk_out[base + j] = kv * kkc[j] * inv;
        k_out[base + j]  = kv * (1.f + (av - 1.f) * kac[j]);
    }
}

// =====================================================================
// RWKV7 delta-rule scan: one block per (b,h), 64 threads (v index),
// each thread keeps one state row S[i][0..63] in registers.
// =====================================================================
__global__ __launch_bounds__(64)
void scan_kernel(const float* __restrict__ r, const float* __restrict__ dec,
                 const float* __restrict__ k, const float* __restrict__ v,
                 const float* __restrict__ kk, const float* __restrict__ a,
                 float* __restrict__ y)
{
    const int b = blockIdx.x / Hn;
    const int h = blockIdx.x % Hn;
    const int i = threadIdx.x;

    float S[HDn];
    #pragma unroll
    for (int j = 0; j < HDn; ++j) S[j] = 0.f;

    __shared__ float sr[HDn], sd[HDn], sk[HDn], skk[HDn], sa[HDn];

    for (int t = 0; t < Tseq; ++t) {
        size_t base = ((size_t)((size_t)b * Tseq + t) * Hn + h) * HDn;
        sr[i]  = r  [base + i];
        sd[i]  = dec[base + i];
        sk[i]  = k  [base + i];
        skk[i] = kk [base + i];
        sa[i]  = a  [base + i];
        float vi = v[base + i];
        __syncthreads();

        float sav = 0.f;
        #pragma unroll
        for (int j = 0; j < HDn; ++j) sav = fmaf(-S[j], skk[j], sav);

        float yv = 0.f;
        #pragma unroll
        for (int j = 0; j < HDn; ++j) {
            float sj = fmaf(S[j], sd[j], fmaf(sav, skk[j] * sa[j], vi * sk[j]));
            S[j] = sj;
            yv = fmaf(sj, sr[j], yv);
        }
        y[base + i] = yv;
        __syncthreads();
    }
}

// =====================================================================
// per-head GroupNorm + bonus + output gate; emit bf16 for final GEMM
// =====================================================================
__global__ void postnorm_kernel(const float* __restrict__ y,
                                const float* __restrict__ r,
                                const float* __restrict__ k,
                                const float* __restrict__ v,
                                const float* __restrict__ g,
                                const float* __restrict__ r_k,
                                const float* __restrict__ ln_w,
                                const float* __restrict__ ln_b,
                                __bf16* __restrict__ yg)
{
    long long idx = (long long)blockIdx.x * blockDim.x + threadIdx.x;
    if (idx >= BTll * Hn) return;
    size_t base = (size_t)idx * HDn;
    int h = (int)(idx % Hn);
    const float* rkh = r_k + h * HDn;
    const float* lw  = ln_w + h * HDn;
    const float* lb  = ln_b + h * HDn;

    float mean = 0.f;
    for (int j = 0; j < HDn; ++j) mean += y[base + j];
    mean *= (1.f / HDn);
    float var = 0.f;
    for (int j = 0; j < HDn; ++j) { float d = y[base + j] - mean; var = fmaf(d, d, var); }
    var *= (1.f / HDn);
    float inv = rsqrtf(var + EPS_);
    float bonus = 0.f;
    for (int j = 0; j < HDn; ++j) bonus = fmaf(r[base + j] * k[base + j], rkh[j], bonus);
    for (int j = 0; j < HDn; ++j) {
        float yn = (y[base + j] - mean) * inv * lw[j] + lb[j];
        yn = fmaf(bonus, v[base + j], yn);
        yg[base + j] = (__bf16)(yn * g[base + j]);
    }
}

// =====================================================================
// host launcher
// =====================================================================
extern "C" void kernel_launch(void* const* d_in, const int* in_sizes, int n_in,
                              void* d_out, int out_size, void* d_ws, size_t ws_size,
                              hipStream_t stream)
{
    (void)in_sizes; (void)n_in; (void)out_size; (void)ws_size;
    const float* x    = (const float*)d_in[0];
    const float* c_r  = (const float*)d_in[1];
    const float* c_w  = (const float*)d_in[2];
    const float* c_k  = (const float*)d_in[3];
    const float* c_v  = (const float*)d_in[4];
    const float* c_a  = (const float*)d_in[5];
    const float* c_g  = (const float*)d_in[6];
    const float* W_r  = (const float*)d_in[7];
    const float* W_k  = (const float*)d_in[8];
    const float* W_v  = (const float*)d_in[9];
    const float* W_o  = (const float*)d_in[10];
    const float* w0   = (const float*)d_in[11];
    const float* w1   = (const float*)d_in[12];
    const float* w2   = (const float*)d_in[13];
    const float* a0   = (const float*)d_in[14];
    const float* a1   = (const float*)d_in[15];
    const float* a2   = (const float*)d_in[16];
    const float* g1   = (const float*)d_in[17];
    const float* g2   = (const float*)d_in[18];
    const float* k_k  = (const float*)d_in[19];
    const float* k_a  = (const float*)d_in[20];
    const float* r_k  = (const float*)d_in[21];
    const float* ln_w = (const float*)d_in[22];
    const float* ln_b = (const float*)d_in[23];

    const size_t NC = (size_t)BTll * Cdim;        // 8.39M elems
    float* out_ptr = (float*)d_out;               // [BT, C]
    float* v_f     = out_ptr + NC;                // v_first lives in d_out

    char* p = (char*)d_ws;
    auto carve = [&](size_t bytes) -> char* {
        char* q = p; p += (bytes + 255) & ~(size_t)255; return q;
    };
    __bf16* xr_bf = (__bf16*)carve(NC * 2);
    __bf16* xk_bf = (__bf16*)carve(NC * 2);
    __bf16* xv_bf = (__bf16*)carve(NC * 2);
    __bf16* yg_bf = (__bf16*)carve(NC * 2);
    __bf16* Wr_bf = (__bf16*)carve((size_t)Cdim * Cdim * 2);
    __bf16* Wk_bf = (__bf16*)carve((size_t)Cdim * Cdim * 2);
    __bf16* Wv_bf = (__bf16*)carve((size_t)Cdim * Cdim * 2);
    __bf16* Wo_bf = (__bf16*)carve((size_t)Cdim * Cdim * 2);
    float* xw_f  = (float*)carve(NC * 4);
    float* xa_f  = (float*)carve(NC * 4);
    float* xg_f  = (float*)carve(NC * 4);
    float* r_f   = (float*)carve(NC * 4);
    float* k_f   = (float*)carve(NC * 4);
    float* kk_f  = (float*)carve(NC * 4);
    float* dec_f = (float*)carve(NC * 4);
    float* a_f   = (float*)carve(NC * 4);
    float* g_f   = (float*)carve(NC * 4);
    float* y_f   = (float*)carve(NC * 4);
    float* t_w   = (float*)carve((size_t)BTll * DW_ * 4);
    float* t_a   = (float*)carve((size_t)BTll * DA_ * 4);
    float* t_g   = (float*)carve((size_t)BTll * DG_ * 4);

    const int TPB = 256;
    const long long wn = (long long)Cdim * Cdim;

    // 1) weights -> bf16
    cvt_f32_bf16<<<(wn + TPB - 1) / TPB, TPB, 0, stream>>>(W_r, Wr_bf, wn);
    cvt_f32_bf16<<<(wn + TPB - 1) / TPB, TPB, 0, stream>>>(W_k, Wk_bf, wn);
    cvt_f32_bf16<<<(wn + TPB - 1) / TPB, TPB, 0, stream>>>(W_v, Wv_bf, wn);
    cvt_f32_bf16<<<(wn + TPB - 1) / TPB, TPB, 0, stream>>>(W_o, Wo_bf, wn);

    // 2) token shift
    token_shift_kernel<<<(NC + TPB - 1) / TPB, TPB, 0, stream>>>(
        x, c_r, c_w, c_k, c_v, c_a, c_g,
        xr_bf, xk_bf, xv_bf, xw_f, xa_f, xg_f);

    // 3) big WMMA GEMMs: r, k, v (v straight into d_out's v_first half)
    dim3 ggrid((unsigned)(BTll / TILE_M), (unsigned)(Cdim / TILE_N));
    wmma_gemm_bf16<<<ggrid, 256, 0, stream>>>(xr_bf, Wr_bf, r_f, (int)BTll, Cdim, Cdim);
    wmma_gemm_bf16<<<ggrid, 256, 0, stream>>>(xk_bf, Wk_bf, k_f, (int)BTll, Cdim, Cdim);
    wmma_gemm_bf16<<<ggrid, 256, 0, stream>>>(xv_bf, Wv_bf, v_f, (int)BTll, Cdim, Cdim);

    // 4) low-rank chains on f32 WMMA (16x16x4), fused activations
    const unsigned MT = (unsigned)(BTll / 16);    // 512
    wmma_gemm_f32<<<dim3(MT, DW_ / 16), 32, 0, stream>>>(
        xw_f, w1, nullptr, t_w, (int)BTll, Cdim, DW_, 1);           // tanh(xw@w1)
    wmma_gemm_f32<<<dim3(MT, DA_ / 16), 32, 0, stream>>>(
        xa_f, a1, nullptr, t_a, (int)BTll, Cdim, DA_, 0);           // xa@a1
    wmma_gemm_f32<<<dim3(MT, DG_ / 16), 32, 0, stream>>>(
        xg_f, g1, nullptr, t_g, (int)BTll, Cdim, DG_, 2);           // sigmoid(xg@g1)
    wmma_gemm_f32<<<dim3(MT, Cdim / 16), 32, 0, stream>>>(
        t_w, w2, w0, dec_f, (int)BTll, DW_, Cdim, 4);               // decay chain
    wmma_gemm_f32<<<dim3(MT, Cdim / 16), 32, 0, stream>>>(
        t_a, a2, a0, a_f, (int)BTll, DA_, Cdim, 3);                 // sigmoid(a0+..)
    wmma_gemm_f32<<<dim3(MT, Cdim / 16), 32, 0, stream>>>(
        t_g, g2, nullptr, g_f, (int)BTll, DG_, Cdim, 0);            // gate

    // 5) kk normalize + k modulation (in-place on k_f)
    long long nh = BTll * Hn;
    prep_kernel<<<(nh + TPB - 1) / TPB, TPB, 0, stream>>>(k_f, a_f, k_k, k_a, kk_f, k_f);

    // 6) sequential scan over T
    scan_kernel<<<Bsz * Hn, HDn, 0, stream>>>(r_f, dec_f, k_f, v_f, kk_f, a_f, y_f);

    // 7) groupnorm + bonus + gate -> bf16
    postnorm_kernel<<<(nh + TPB - 1) / TPB, TPB, 0, stream>>>(
        y_f, r_f, k_f, v_f, g_f, r_k, ln_w, ln_b, yg_bf);

    // 8) output GEMM -> d_out
    wmma_gemm_bf16<<<ggrid, 256, 0, stream>>>(yg_bf, Wo_bf, out_ptr, (int)BTll, Cdim, Cdim);
}
